// MultiScaleAttentionPE_55250459296238
// MI455X (gfx1250) — compile-verified, matrix operand order checked by reading, not run on hardware
//
#include <hip/hip_runtime.h>

typedef _Float16 half_t;
typedef _Float16 v16h __attribute__((ext_vector_type(16)));
typedef _Float16 v8h  __attribute__((ext_vector_type(8)));
typedef float    v8f  __attribute__((ext_vector_type(8)));

#define CDIM 128
#define N0 262144
#define N1 65536
#define N2 16384
#define N3 4096
#define N4 1024

#if defined(__has_builtin)
#if __has_builtin(__builtin_amdgcn_sched_barrier)
#define SCHED_BARRIER() __builtin_amdgcn_sched_barrier(0)
#endif
#endif
#ifndef SCHED_BARRIER
#define SCHED_BARRIER()
#endif

// Pt[n][k] = (f16) P[k][n] ; P is (256 x 128) f32 row-major, Pt is (128 x 256) f16
__global__ __launch_bounds__(256) void transpose_P_kernel(const float* __restrict__ P,
                                                          half_t* __restrict__ Pt) {
    int id = blockIdx.x * 256 + threadIdx.x;   // 0 .. 32767
    int n = id >> 8;                           // 0..127
    int k = id & 255;                          // 0..255
    Pt[n * 256 + k] = (half_t)P[k * CDIM + n];
}

// feat4 = f0[:N4] = (xyz0 @ W_all + b_all)[:N4]   (sliced, NOT gathered)
__global__ __launch_bounds__(256) void feat4_kernel(const float* __restrict__ xyz0,
                                                    const float* __restrict__ W,
                                                    const float* __restrict__ b,
                                                    float* __restrict__ out) {
    int id = blockIdx.x * 256 + threadIdx.x;   // 0 .. N4*C-1
    int i = id >> 7;
    int c = id & 127;
    float x = xyz0[i * 3 + 0], y = xyz0[i * 3 + 1], z = xyz0[i * 3 + 2];
    out[id] = x * W[c] + y * W[CDIM + c] + z * W[2 * CDIM + c] + b[c];
}

union frag_t { v16h v; v8h h[2]; };

// One level: out = concat([feat_c[knn] + (xyz_f - xyz_c[knn]) @ Wd + bd, f0_fine]) @ P + pb
// Block = 256 threads (8 waves) -> 64 fine points x 128 output channels.
// Each wave owns 16 output columns and 4 M-tiles. The K loop is software-pipelined
// with VGPR double-buffering; amdgpu_waves_per_eu(2) frees enough registers for both
// buffers to stay live, and sched_barrier keeps the (k+1) loads ahead of the k WMMAs.
__global__ __launch_bounds__(256)
__attribute__((amdgpu_waves_per_eu(2)))
void level_kernel(const float* __restrict__ xyz0,
                  const int*   __restrict__ idx_f,   // null => identity
                  const int*   __restrict__ idx_c,
                  const int*   __restrict__ knn,
                  const float* __restrict__ featc,   // (Nc x C) f32
                  const float* __restrict__ Wall,    // (3 x C)
                  const float* __restrict__ ball,    // (C)
                  const float* __restrict__ Wd,      // (3 x C)
                  const float* __restrict__ bd,      // (C)
                  const half_t* __restrict__ Pt,     // (C x 2C) f16 transposed
                  const float* __restrict__ pb,      // (C)
                  float* __restrict__ out)           // (Nf x C)
{
    // A tile: 64 rows x 256 K, f16. Stride 264 halves = 528 B (16B-aligned, bank-spread).
    __shared__ half_t Asm[64][264];

    const int tile = blockIdx.x;
    const int t    = threadIdx.x;

    // ---------- Phase 1: build A = [nei | f_f] for 64 points (4 rows per thread) ----------
    {
        const int g8 = (t & 15) << 3;   // channel group base (0,8,...,120)
#pragma unroll
        for (int u4 = 0; u4 < 4; ++u4) {
            const int p = (t >> 4) + (u4 << 4);   // point in tile (0..63)
            const int i = tile * 64 + p;          // fine point index

            const int fg = idx_f ? idx_f[i] : i;  // gathered fine position (for delta)
            const int j  = knn[i];                // coarse local index
            const int cg = idx_c[j];              // gathered coarse position

            const float xf = xyz0[fg * 3 + 0], yf = xyz0[fg * 3 + 1], zf = xyz0[fg * 3 + 2];
            const float dx = xf - xyz0[cg * 3 + 0];
            const float dy = yf - xyz0[cg * 3 + 1];
            const float dz = zf - xyz0[cg * 3 + 2];
            // f_f row i comes from raw xyz0 row i (reference slices f0, not gathers)
            const float xr = xyz0[i * 3 + 0], yr = xyz0[i * 3 + 1], zr = xyz0[i * 3 + 2];

            const float* fc = featc + (size_t)j * CDIM + g8;
            v8h nv, fv;
#pragma unroll
            for (int u = 0; u < 8; ++u) {
                const int c = g8 + u;
                float nei = fc[u] + dx * Wd[c] + dy * Wd[CDIM + c] + dz * Wd[2 * CDIM + c] + bd[c];
                float ff  = xr * Wall[c] + yr * Wall[CDIM + c] + zr * Wall[2 * CDIM + c] + ball[c];
                nv[u] = (half_t)nei;
                fv[u] = (half_t)ff;
            }
            *(v8h*)&Asm[p][g8]        = nv;   // 16B store, aligned
            *(v8h*)&Asm[p][CDIM + g8] = fv;
        }
    }
    __syncthreads();

    // ---------- Phase 2: pipelined K loop; 4 M-tiles x 16 cols per wave, K = 8 x 32 ----------
    const int lane = t & 31;
    const int wave = t >> 5;                 // output columns [16*wave, 16*wave+16)
    const int r    = lane & 15;              // A row / C column-in-tile
    const int kh   = (lane >> 4) << 3;       // 0 or 8 (K sub-offset per 16-bit frag layout)
    const int coln = (wave << 4) + r;        // global output column for B/C

    const half_t* pbase = Pt + (size_t)coln * 256 + kh;   // + kb walks K steps

    frag_t bm[2];
    frag_t am[2][4];

    // prologue: load k-step 0
    bm[0].h[0] = *(const v8h*)(pbase);
    bm[0].h[1] = *(const v8h*)(pbase + 16);
#pragma unroll
    for (int mt = 0; mt < 4; ++mt) {
        const int row = (mt << 4) + r;
        am[0][mt].h[0] = *(const v8h*)&Asm[row][kh];
        am[0][mt].h[1] = *(const v8h*)&Asm[row][16 + kh];
    }

    v8f acc[4] = {v8f{}, v8f{}, v8f{}, v8f{}};
#pragma unroll
    for (int kk = 0; kk < 8; ++kk) {
        const int cur = kk & 1;
        const int nxt = cur ^ 1;
        if (kk < 7) {
            const int kb = (kk + 1) << 5;
            bm[nxt].h[0] = *(const v8h*)(pbase + kb);
            bm[nxt].h[1] = *(const v8h*)(pbase + kb + 16);
#pragma unroll
            for (int mt = 0; mt < 4; ++mt) {
                const int row = (mt << 4) + r;
                am[nxt][mt].h[0] = *(const v8h*)&Asm[row][kb + kh];
                am[nxt][mt].h[1] = *(const v8h*)&Asm[row][kb + 16 + kh];
            }
        }
        // Keep the (k+1) prefetch group issued before the k-th WMMA group.
        SCHED_BARRIER();
#pragma unroll
        for (int mt = 0; mt < 4; ++mt) {
            acc[mt] = __builtin_amdgcn_wmma_f32_16x16x32_f16(
                /*neg_a=*/false, am[cur][mt].v, /*neg_b=*/false, bm[cur].v,
                /*c_mod=*/(short)0, acc[mt], /*reuse_a=*/false, /*reuse_b=*/false);
        }
        SCHED_BARRIER();
    }

    // ---------- Phase 3: bias + store (C/D layout: lane=col, VGPR u -> row u + 8*(lane>=16)) ----------
    const int mbase = (lane >> 4) << 3;
    const float pbv = pb[coln];
#pragma unroll
    for (int mt = 0; mt < 4; ++mt) {
        const int rowbase = tile * 64 + (mt << 4) + mbase;
#pragma unroll
        for (int u = 0; u < 8; ++u) {
            out[(size_t)(rowbase + u) * CDIM + coln] = acc[mt][u] + pbv;
        }
    }
}

extern "C" void kernel_launch(void* const* d_in, const int* in_sizes, int n_in,
                              void* d_out, int out_size, void* d_ws, size_t ws_size,
                              hipStream_t stream) {
    const float* xyz0  = (const float*)d_in[0];
    const int*   idx4  = (const int*)d_in[1];
    const int*   idx3  = (const int*)d_in[2];
    const int*   idx2  = (const int*)d_in[3];
    const int*   idx1  = (const int*)d_in[4];
    const int*   knn34 = (const int*)d_in[5];
    const int*   knn23 = (const int*)d_in[6];
    const int*   knn12 = (const int*)d_in[7];
    const int*   knn01 = (const int*)d_in[8];
    const float* W_all = (const float*)d_in[9];
    const float* b_all = (const float*)d_in[10];
    const float* W3 = (const float*)d_in[11]; const float* b3 = (const float*)d_in[12];
    const float* W2 = (const float*)d_in[13]; const float* b2 = (const float*)d_in[14];
    const float* W1 = (const float*)d_in[15]; const float* b1 = (const float*)d_in[16];
    const float* W0 = (const float*)d_in[17]; const float* b0 = (const float*)d_in[18];
    const float* P3 = (const float*)d_in[19]; const float* pb3 = (const float*)d_in[20];
    const float* P2 = (const float*)d_in[21]; const float* pb2 = (const float*)d_in[22];
    const float* P1 = (const float*)d_in[23]; const float* pb1 = (const float*)d_in[24];
    const float* P0 = (const float*)d_in[25]; const float* pb0 = (const float*)d_in[26];

    float* out   = (float*)d_out;
    float* feat4 = out;                       // N4*C
    float* feat3 = feat4 + (size_t)N4 * CDIM; // N3*C
    float* feat2 = feat3 + (size_t)N3 * CDIM; // N2*C
    float* feat1 = feat2 + (size_t)N2 * CDIM; // N1*C
    float* feat0 = feat1 + (size_t)N1 * CDIM; // N0*C

    half_t* Pt3 = (half_t*)d_ws;              // each 128*256 f16 = 64KB
    half_t* Pt2 = Pt3 + CDIM * 2 * CDIM;
    half_t* Pt1 = Pt2 + CDIM * 2 * CDIM;
    half_t* Pt0 = Pt1 + CDIM * 2 * CDIM;

    const int tpB = 256;
    const int trB = (CDIM * 2 * CDIM) / tpB;  // 128 blocks
    transpose_P_kernel<<<trB, tpB, 0, stream>>>(P3, Pt3);
    transpose_P_kernel<<<trB, tpB, 0, stream>>>(P2, Pt2);
    transpose_P_kernel<<<trB, tpB, 0, stream>>>(P1, Pt1);
    transpose_P_kernel<<<trB, tpB, 0, stream>>>(P0, Pt0);

    feat4_kernel<<<(N4 * CDIM) / tpB, tpB, 0, stream>>>(xyz0, W_all, b_all, feat4);

    // Levels are sequentially dependent; stream ordering enforces it.
    level_kernel<<<N3 / 64, tpB, 0, stream>>>(xyz0, idx3, idx4, knn34, feat4,
                                              W_all, b_all, W3, b3, Pt3, pb3, feat3);
    level_kernel<<<N2 / 64, tpB, 0, stream>>>(xyz0, idx2, idx3, knn23, feat3,
                                              W_all, b_all, W2, b2, Pt2, pb2, feat2);
    level_kernel<<<N1 / 64, tpB, 0, stream>>>(xyz0, idx1, idx2, knn12, feat2,
                                              W_all, b_all, W1, b1, Pt1, pb1, feat1);
    level_kernel<<<N0 / 64, tpB, 0, stream>>>(xyz0, nullptr, idx1, knn01, feat1,
                                              W_all, b_all, W0, b0, Pt0, pb0, feat0);
}
